// DREAMCell_52106543235367
// MI455X (gfx1250) — matrix operand order repaired and checked by reading, not compiled, not run on hardware
//
#include <hip/hip_runtime.h>
#include <hip/hip_bf16.h>

// ---------------- problem constants ----------------
#define NWG  16
#define TBLK 256
#define BSZ  16
#define TT   256
#define II   512
#define HHH  1024
#define RRR  16

// padded LDS row strides (bf16 elements): +8 elems = 16B = 4-bank skew per row
#define CSTRIDE 1032     // K=1024 rows (C slice, h panel)
#define WSTRIDE 520      // K=512 rows (W/Bm slices, err/xu panels)

typedef __attribute__((ext_vector_type(16))) __bf16   v16bf;
typedef __attribute__((ext_vector_type(8)))  float    v8f;
typedef __attribute__((ext_vector_type(4)))  unsigned uv4;

struct Q2 { uv4 lo, hi; };

__device__ __forceinline__ unsigned short f2bf(float f) {
    unsigned u = __float_as_uint(f);
    u += 0x7FFFu + ((u >> 16) & 1u);   // round-to-nearest-even
    return (unsigned short)(u >> 16);
}

// A fragment: 16x32 bf16 row-major. lane<16: K = kb+0..7 (lo), kb+16..23 (hi); lane>=16: +8.
__device__ __forceinline__ v16bf load_a(const unsigned short* row, int kb, int sel) {
    Q2 q;
    q.lo = *reinterpret_cast<const uv4*>(row + kb + 8 * sel);
    q.hi = *reinterpret_cast<const uv4*>(row + kb + 16 + 8 * sel);
    return __builtin_bit_cast(v16bf, q);
}
// B fragment: 32x16 bf16; K-run contiguous in a row-major weight row.
// lane<16: K = kb+0..7 (lo), kb+16..23 (hi); lane>=16: +8.
__device__ __forceinline__ v16bf load_b(const unsigned short* row, int kb, int sel) {
    const unsigned short* p = row + kb + 8 * sel;
    Q2 q;
    q.lo = *reinterpret_cast<const uv4*>(p);
    q.hi = *reinterpret_cast<const uv4*>(p + 16);
    return __builtin_bit_cast(v16bf, q);
}

// 8-step GEMM entirely out of LDS (ds_load_b128 operands, ~LDS latency only).
__device__ __forceinline__ v8f gemm8(const unsigned short* arow, const unsigned short* brow,
                                     int kb0, int sel, v8f acc) {
#pragma unroll
    for (int ks = 0; ks < 8; ++ks) {
        int kb = kb0 + ks * 32;
        v16bf a = load_a(arow, kb, sel);
        v16bf b = load_b(brow, kb, sel);
        acc = __builtin_amdgcn_wmma_f32_16x16x32_bf16(false, a, false, b,
                                                      (short)0, acc, false, false);
    }
    return acc;
}

// ---------------- device-scope split barrier ----------------
__device__ __forceinline__ void gbar_arrive(unsigned* bar, unsigned* bexp) {
    __syncthreads();
    __builtin_amdgcn_fence(__ATOMIC_RELEASE, "agent");
    if (threadIdx.x == 0) {
        unsigned old = __hip_atomic_fetch_add(&bar[0], 1u, __ATOMIC_ACQ_REL, __HIP_MEMORY_SCOPE_AGENT);
        if (old == NWG - 1) {
            __hip_atomic_store(&bar[0], 0u, __ATOMIC_RELAXED, __HIP_MEMORY_SCOPE_AGENT);
            __hip_atomic_fetch_add(&bar[1], 1u, __ATOMIC_RELEASE, __HIP_MEMORY_SCOPE_AGENT);
        }
    }
    (*bexp)++;
}
__device__ __forceinline__ void gbar_wait(unsigned* bar, unsigned target) {
    if (threadIdx.x == 0) {
        while (__hip_atomic_load(&bar[1], __ATOMIC_ACQUIRE, __HIP_MEMORY_SCOPE_AGENT) < target)
            __builtin_amdgcn_s_sleep(1);
    }
    __syncthreads();
    __builtin_amdgcn_fence(__ATOMIC_ACQUIRE, "agent");
}

// ---------------- workspace layout (bytes) ----------------
static constexpr size_t OFF_BAR  = 0;                       // 2 uints
static constexpr size_t OFF_HU   = 1024;                    // 16x16 f32 (hU)
static constexpr size_t OFF_ENP  = 2048;                    // 16x16 f32 (err-norm partials)
static constexpr size_t OFF_EVP  = 3072;                    // 16x16 f32 (error_var partials)
static constexpr size_t OFF_HF0  = 4096;                    // h f32 ping (64KB)
static constexpr size_t OFF_HF1  = OFF_HF0 + 65536;         // h f32 pong
static constexpr size_t OFF_HB0  = OFF_HF1 + 65536;         // h bf16 ping (32KB)
static constexpr size_t OFF_HB1  = OFF_HB0 + 32768;
static constexpr size_t OFF_ERRF = OFF_HB1 + 32768;         // error f32 (32KB)
static constexpr size_t OFF_ERRB = OFF_ERRF + 32768;        // error bf16 (16KB)
static constexpr size_t OFF_XUB  = OFF_ERRB + 16384;        // x_unit bf16 (16KB)
static constexpr size_t OFF_CBF  = OFF_XUB + 16384;         // C bf16 (1MB)
static constexpr size_t OFF_WBF  = OFF_CBF + 1048576;       // W bf16 (1MB)
static constexpr size_t OFF_BMBF = OFF_WBF + 1048576;       // Bm bf16 (1MB)

// dynamic LDS: 20336 floats + (33024 + 33280 + 33280 + 16640) ushorts = 313792 B
static constexpr size_t LDS_BYTES = 20336u * 4u + (33024u + 33280u + 33280u + 16640u) * 2u;

// ---------------- init: f32 -> bf16 weights + barrier reset ----------------
__global__ void dream_init(const float* __restrict__ C, const float* __restrict__ W,
                           const float* __restrict__ Bm,
                           unsigned short* __restrict__ Cb, unsigned short* __restrict__ Wb,
                           unsigned short* __restrict__ Bmb, unsigned* __restrict__ bar) {
    int i = blockIdx.x * blockDim.x + threadIdx.x;
    if (i < II * HHH) {
        Cb[i]  = f2bf(C[i]);
        Wb[i]  = f2bf(W[i]);
        Bmb[i] = f2bf(Bm[i]);
    }
    if (i == 0) { bar[0] = 0u; bar[1] = 0u; }
}

// ---------------- persistent scan kernel ----------------
__global__ void __launch_bounds__(TBLK, 1)
dream_scan(const float* __restrict__ x, const float* __restrict__ V,
           const float* __restrict__ p_eta, const float* __restrict__ p_tau,
           const float* __restrict__ p_ltc,
           const float* __restrict__ h0, const float* __restrict__ U0,
           const float* __restrict__ ev0, const float* __restrict__ at0,
           float* __restrict__ out, unsigned* __restrict__ bar,
           float* __restrict__ ghU, float* __restrict__ gEnP, float* __restrict__ gEvP,
           float* __restrict__ hf0, float* __restrict__ hf1,
           unsigned short* __restrict__ hb0, unsigned short* __restrict__ hb1,
           float* __restrict__ gErrF, unsigned short* __restrict__ gErrB,
           unsigned short* __restrict__ gXuB,
           const unsigned short* __restrict__ Cb, const unsigned short* __restrict__ Wb,
           const unsigned short* __restrict__ Bmb) {
    const int g    = blockIdx.x;     // workgroup id == batch owned for U / N-slice id
    const int tid  = threadIdx.x;
    const int lane = tid & 31;
    const int wv   = tid >> 5;       // 8 wave32s
    const int l15  = lane & 15;
    const int sel  = lane >> 4;
    unsigned bexp  = 0;

    extern __shared__ float smem[];
    float* sU   = smem;              // 16384 : U[g]  (h*16+r)
    float* sG   = sU  + 16384;       //  2048 : WMMA partial tiles
    float* sH   = sG  + 2048;        //  1024 : h row of batch g (f32)
    float* sRed = sH  + 1024;        //   256 : reductions
    float* sXn  = sRed + 256;        //    16 : ||x_t||
    float* sEvm = sXn  + 16;         //    16 : mean(error_var)
    float* sAt  = sEvm + 16;         //    16 : adaptive_tau (persistent)
    float* sSur = sAt  + 16;         //    16 : surprise
    float* sA   = sSur + 16;         //    16 : leak a
    float* sEVr = sA   + 16;         //    16 : error @ V  (batch g)
    float* sEn  = sEVr + 16;         //    16 : ||error||^2 partial
    float* sEVs = sEn  + 16;         //   512 : error_var slice [16b x 32i]
    unsigned short* sCw  = (unsigned short*)(sEVs + 512);  // 32 x CSTRIDE  (C slice)
    unsigned short* sWw  = sCw + 32 * CSTRIDE;             // 64 x WSTRIDE  (W slice)
    unsigned short* sBmw = sWw + 64 * WSTRIDE;             // 64 x WSTRIDE  (Bm slice)
    unsigned short* sAct = sBmw + 64 * WSTRIDE;            // staging: h (16xCSTRIDE) or err+xu (2x16xWSTRIDE)
    unsigned* sCwU  = (unsigned*)sCw;
    unsigned* sWwU  = (unsigned*)sWw;
    unsigned* sBmwU = (unsigned*)sBmw;
    unsigned* sActU = (unsigned*)sAct;

    const float eta     = *p_eta;
    const float tau_sys = *p_tau;
    const float sscale  = __expf(*p_ltc);

    // ---- loop-invariant GEMM geometry (LDS row pointers) ----
    const int p1_ntile = wv & 1, p1_kb0 = (wv >> 1) * 256;   // P1b: 2 N-tiles x 4 K-chunks
    const int p2_ntile = wv & 3, p2_kb0 = (wv >> 2) * 256;   // P2c: 4 N-tiles x 2 K-halves
    const unsigned short* p1_arow = sAct + (size_t)l15 * CSTRIDE;
    const unsigned short* p1_brow = sCw + (size_t)(p1_ntile * 16 + l15) * CSTRIDE;
    const unsigned short* p2_aE   = sAct + (size_t)l15 * WSTRIDE;
    const unsigned short* p2_aX   = sAct + (size_t)(16 + l15) * WSTRIDE;
    const unsigned short* p2_bW   = sWw + (size_t)(p2_ntile * 16 + l15) * WSTRIDE;
    const unsigned short* p2_bB   = sBmw + (size_t)(p2_ntile * 16 + l15) * WSTRIDE;

    // ---- stage weight slices into LDS (once) ----
    {
        const unsigned* CbU  = (const unsigned*)(Cb + (size_t)g * 32 * HHH);
        const unsigned* WbU  = (const unsigned*)(Wb + (size_t)g * 64 * II);
        const unsigned* BmbU = (const unsigned*)(Bmb + (size_t)g * 64 * II);
        for (int k = 0; k < 64; ++k) {               // 16384 uints (32 rows x 512)
            int e = k * 256 + tid, row = e >> 9, colp = e & 511;
            sCwU[row * (CSTRIDE / 2) + colp] = CbU[row * 512 + colp];
        }
        for (int k = 0; k < 64; ++k) {               // 16384 uints (64 rows x 256)
            int e = k * 256 + tid, row = e >> 8, colp = e & 255;
            sWwU[row * (WSTRIDE / 2) + colp]  = WbU[row * 256 + colp];
            sBmwU[row * (WSTRIDE / 2) + colp] = BmbU[row * 256 + colp];
        }
    }

    // ---- initial state ----
    for (int k = 0; k < 64; ++k) sU[k * 256 + tid] = U0[g * (HHH * RRR) + k * 256 + tid];
    for (int k = 0; k < 2; ++k) {                    // error_var slice: i in [32g, 32g+32)
        int idx = k * 256 + tid, b = idx >> 5, ii = idx & 31;
        sEVs[idx] = ev0[b * II + g * 32 + ii];
    }
    for (int k = 0; k < 4; ++k) {
        int j = k * 256 + tid;
        float hv = h0[g * HHH + j];
        hf0[g * HHH + j] = hv;
        hb0[g * HHH + j] = f2bf(hv);
    }
    if (tid < 16) sAt[tid] = at0[tid];
    gbar_arrive(bar, &bexp);
    gbar_wait(bar, bexp);

    for (int t = 0; t < TT; ++t) {
        const float*          hcf = (t & 1) ? hf1 : hf0;
        float*                hnf = (t & 1) ? hf0 : hf1;
        const unsigned*       hcbU = (const unsigned*)((t & 1) ? hb1 : hb0);
        unsigned short*       hnb = (t & 1) ? hb0 : hb1;

        // ---- stage h panel (16x1024 bf16) into sAct: bulk copy, full MLP ----
        for (int k = 0; k < 32; ++k) {               // 8192 uints (16 rows x 512)
            int e = k * 256 + tid, row = e >> 9, colp = e & 511;
            sActU[row * (CSTRIDE / 2) + colp] = hcbU[row * 512 + colp];
        }

        // ---- P0: ||x_t|| (all b, redundant per WG) ----
        {
            int b = tid >> 4, l = tid & 15;
            const float* xb = x + ((size_t)b * TT + t) * II;
            float xs = 0.f;
            for (int k = 0; k < 32; ++k) { float v = xb[l + 16 * k]; xs += v * v; }
            sRed[tid] = xs; __syncthreads();
            if (tid < 16) { float s = 0.f; for (int j = 0; j < 16; ++j) s += sRed[tid * 16 + j]; sXn[tid] = sqrtf(s); }
            __syncthreads();
        }

        // ---- P1a: h row g (f32); hU[g,:]; error_var partial sums -> global ----
        for (int k = 0; k < 4; ++k) sH[k * 256 + tid] = hcf[g * HHH + k * 256 + tid];
        __syncthreads();
        {
            int r = tid & 15, part = tid >> 4;
            float s = 0.f;
            for (int j = 0; j < 64; ++j) { int h = part * 64 + j; s += sH[h] * sU[h * 16 + r]; }
            sRed[tid] = s; __syncthreads();
            if (tid < 16) {
                float v = 0.f;
                for (int p = 0; p < 16; ++p) v += sRed[p * 16 + tid];
                ghU[g * 16 + tid] = v;
                float es = 0.f;
                for (int j = 0; j < 32; ++j) es += sEVs[tid * 32 + j];   // OLD error_var slice
                gEvP[g * 16 + tid] = es;
            }
        }
        gbar_arrive(bar, &bexp);          // B1 arrive (hidden under GEMM)
        const unsigned b1 = bexp;

        // ---- P1b: x_pred GEMM slice: M=16(b), N=32 (i in [32g,32g+32)), K=1024, all-LDS ----
        {
            v8f acc = {0.f, 0.f, 0.f, 0.f, 0.f, 0.f, 0.f, 0.f};
            acc = gemm8(p1_arow, p1_brow, p1_kb0, sel, acc);
            int pi = p1_ntile * 4 + (p1_kb0 >> 8);
            for (int j = 0; j < 8; ++j) sG[pi * 256 + lane * 8 + j] = acc[j];
        }
        if (tid < 16) sEn[tid] = 0.f;
        gbar_wait(bar, b1);               // B1 done: all WGs' hU visible; also syncs LDS

        // ---- P1b epilogue: error, x_unit, err-norm partials ----
        for (int q = 0; q < 2; ++q) {
            int o = q * 256 + tid;
            int ntile = o >> 8, rem = o & 255, m = rem >> 4, n = rem & 15;
            int i = g * 32 + ntile * 16 + n;
            int L = n + 16 * (m >> 3), j = m & 7;
            float acc = 0.f;
            for (int kp = 0; kp < 4; ++kp) acc += sG[(ntile * 4 + kp) * 256 + L * 8 + j];
            float rk = 0.f;
            for (int r = 0; r < 16; ++r) rk += ghU[m * 16 + r] * V[i * 16 + r];
            float xn = sXn[m];
            float xp = tanhf(acc + rk) * xn;
            float xv = x[((size_t)m * TT + t) * II + i];
            float err = xv - xp;
            gErrF[m * II + i] = err;
            gErrB[m * II + i] = f2bf(err);
            float xu = xv / (xn + 1e-6f);
            xu = fminf(fmaxf(xu, -1.f), 1.f);
            gXuB[m * II + i] = f2bf(xu);
            atomicAdd(&sEn[m], err * err);
        }
        __syncthreads();
        if (tid < 16) gEnP[g * 16 + tid] = sEn[tid];
        gbar_arrive(bar, &bexp);          // B2: error visible everywhere
        gbar_wait(bar, bexp);

        // ---- stage err + xu panels (16x512 bf16 each) into sAct ----
        {
            const unsigned* gErrBU = (const unsigned*)gErrB;
            const unsigned* gXuBU  = (const unsigned*)gXuB;
            for (int k = 0; k < 16; ++k) {           // 4096 uints each (16 rows x 256)
                int e = k * 256 + tid, row = e >> 8, colp = e & 255;
                sActU[row * (WSTRIDE / 2) + colp]                      = gErrBU[row * 256 + colp];
                sActU[(16 + row) * (WSTRIDE / 2) + colp]               = gXuBU[row * 256 + colp];
            }
        }

        // ---- P2a: per-batch scalar chain (redundant, identical across WGs) ----
        if (tid < 16) {
            int b = tid;
            float en2 = 0.f, evs = 0.f;
            for (int w2 = 0; w2 < 16; ++w2) { en2 += gEnP[w2 * 16 + b]; evs += gEvP[w2 * 16 + b]; }
            float rel = sqrtf(en2) / (sXn[b] + 1e-6f);
            float ent = 0.5f * __logf(17.0794684f * (evs * (1.f / 512.f) + 1e-6f));  // log(2*pi*e*v)
            ent = fminf(fmaxf(ent, 0.f), 2.f);
            float at = fminf(0.999f * sAt[b] + 0.001f * rel, 0.8f);
            sAt[b] = at;
            float efft = 0.3f * (0.5f * (1.f + 0.1f * ent)) + 0.7f * at;
            float sur = 1.f / (1.f + __expf(-(rel - efft) * 10.f));       // /GAMMA
            sSur[b] = sur;
            float taueff = tau_sys / (1.f + sur * sscale);
            taueff = fminf(fmaxf(taueff, 0.01f), 50.f);
            float a = 0.1f / (taueff + 0.1f);                             // DT/(tau+DT)
            sA[b] = fminf(fmaxf(a, 0.01f), 0.5f);
        }
        __syncthreads();

        // ---- P2b: eV = error[g] @ V; U[g] update; error_var slice update ----
        {
            int r = tid & 15, part = tid >> 4;
            float s = 0.f;
            for (int j = 0; j < 32; ++j) { int i = part * 32 + j; s += gErrF[g * II + i] * V[i * 16 + r]; }
            sRed[tid] = s; __syncthreads();
            if (tid < 16) { float v = 0.f; for (int p = 0; p < 16; ++p) v += sRed[p * 16 + tid]; sEVr[tid] = v; }
            __syncthreads();
        }
        {
            float sur = sSur[g];
            float lam = 0.01f * (1.f + sur);
            float es  = eta * sur;
            float ss  = 0.f;
            for (int k = 0; k < 64; ++k) {
                int idx = k * 256 + tid;
                int hI = idx >> 4, r = idx & 15;
                float u  = sU[idx];
                float un = u + (-lam * u + es * sH[hI] * sEVr[r]) * 0.1f;
                sU[idx] = un;
                ss += un * un;
            }
            sRed[tid] = ss; __syncthreads();
            for (int s2 = 128; s2 > 0; s2 >>= 1) { if (tid < s2) sRed[tid] += sRed[tid + s2]; __syncthreads(); }
            float factor = fminf(1.f / (sqrtf(sRed[0]) + 1e-6f), 1.5f);
            __syncthreads();
            for (int k = 0; k < 64; ++k) sU[k * 256 + tid] *= factor;
        }
        // error_var slice update (WG-owned columns [32g, 32g+32))
        for (int k = 0; k < 2; ++k) {
            int idx = k * 256 + tid, b = idx >> 5, ii = idx & 31;
            float e = gErrF[b * II + g * 32 + ii];
            sEVs[idx] = 0.9f * sEVs[idx] + 0.1f * e * e;
        }
        __syncthreads();

        // ---- P2c: input_effect GEMM slice: M=16, N=64 (hh in [64g,64g+64)), K=512, all-LDS ----
        {
            v8f acc = {0.f, 0.f, 0.f, 0.f, 0.f, 0.f, 0.f, 0.f};
            acc = gemm8(p2_aE, p2_bW, p2_kb0, sel, acc);  // KAPPA * error @ W^T (KAPPA=1)
            acc = gemm8(p2_aX, p2_bB, p2_kb0, sel, acc);  // + x_unit @ Bm^T
            int pi = p2_ntile * 2 + (p2_kb0 >> 8);
            for (int j = 0; j < 8; ++j) sG[pi * 256 + lane * 8 + j] = acc[j];
        }
        __syncthreads();

        // ---- P2c epilogue: h_new, write outputs ----
        for (int q = 0; q < 4; ++q) {
            int o = q * 256 + tid;
            int ntile = o >> 8, rem = o & 255, m = rem >> 4, n = rem & 15;
            int hh = g * 64 + ntile * 16 + n;
            int L = n + 16 * (m >> 3), j = m & 7;
            float ie = sG[(ntile * 2) * 256 + L * 8 + j] + sG[(ntile * 2 + 1) * 256 + L * 8 + j];
            float th = tanhf(ie);
            float a = sA[m];
            float hold = hcf[(size_t)m * HHH + hh];
            float hn = (1.f - a) * hold + a * th;
            hnf[(size_t)m * HHH + hh] = hn;
            hnb[(size_t)m * HHH + hh] = f2bf(hn);
            out[((size_t)m * TT + t) * HHH + hh] = hn;
        }
        gbar_arrive(bar, &bexp);          // B3: step complete, h_new visible
        gbar_wait(bar, bexp);
    }
}

// ---------------- host launcher ----------------
extern "C" void kernel_launch(void* const* d_in, const int* in_sizes, int n_in,
                              void* d_out, int out_size, void* d_ws, size_t ws_size,
                              hipStream_t stream) {
    (void)in_sizes; (void)n_in; (void)out_size; (void)ws_size;
    const float* x   = (const float*)d_in[0];
    const float* C   = (const float*)d_in[1];
    const float* W   = (const float*)d_in[2];
    const float* Bm  = (const float*)d_in[3];
    const float* V   = (const float*)d_in[4];
    const float* eta = (const float*)d_in[5];
    const float* tau = (const float*)d_in[6];
    const float* ltc = (const float*)d_in[7];
    const float* h0  = (const float*)d_in[8];
    const float* U0  = (const float*)d_in[9];
    const float* ev0 = (const float*)d_in[10];
    const float* at0 = (const float*)d_in[11];

    char* ws = (char*)d_ws;
    unsigned*       bar   = (unsigned*)(ws + OFF_BAR);
    float*          ghU   = (float*)(ws + OFF_HU);
    float*          gEnP  = (float*)(ws + OFF_ENP);
    float*          gEvP  = (float*)(ws + OFF_EVP);
    float*          hf0   = (float*)(ws + OFF_HF0);
    float*          hf1   = (float*)(ws + OFF_HF1);
    unsigned short* hb0   = (unsigned short*)(ws + OFF_HB0);
    unsigned short* hb1   = (unsigned short*)(ws + OFF_HB1);
    float*          gErrF = (float*)(ws + OFF_ERRF);
    unsigned short* gErrB = (unsigned short*)(ws + OFF_ERRB);
    unsigned short* gXuB  = (unsigned short*)(ws + OFF_XUB);
    unsigned short* Cb    = (unsigned short*)(ws + OFF_CBF);
    unsigned short* Wb    = (unsigned short*)(ws + OFF_WBF);
    unsigned short* Bmb   = (unsigned short*)(ws + OFF_BMBF);

    dream_init<<<dim3((II * HHH + 255) / 256), dim3(256), 0, stream>>>(C, W, Bm, Cb, Wb, Bmb, bar);

    dream_scan<<<dim3(NWG), dim3(TBLK), LDS_BYTES, stream>>>(
        x, V, eta, tau, ltc, h0, U0, ev0, at0,
        (float*)d_out, bar, ghU, gEnP, gEvP, hf0, hf1, hb0, hb1,
        gErrF, gErrB, gXuB, Cb, Wb, Bmb);
}
